// PointNetFeaturePropagation_523986010541
// MI455X (gfx1250) — compile-verified, matrix operand order checked by reading, not run on hardware
//
#include <hip/hip_runtime.h>
#include <hip/hip_bf16.h>
#include <float.h>

typedef __attribute__((ext_vector_type(16))) _Float16 v16h;
typedef __attribute__((ext_vector_type(8)))  _Float16 v8h;
typedef __attribute__((ext_vector_type(8)))  float    v8f;

#define B_   8
#define NPT  8192
#define SPT  2048
#define C1_  128
#define C2_  256
#define CIN  384
#define O_   256
#define T_   256
#define ROWS (B_ * NPT)   // 65536 (b,n) rows

__device__ __forceinline__ float gelu_exact(float x) {
    return 0.5f * x * (1.0f + erff(x * 0.7071067811865475f));
}

// ---------------------------------------------------------------------------
// 1) 3-NN: xyz2 for one batch staged entirely in LDS (24KB), one thread per
//    query point keeps a register top-3, writes indices + normalized weights.
// ---------------------------------------------------------------------------
__global__ void knn_kernel(const float* __restrict__ xyz1,
                           const float* __restrict__ xyz2,
                           int* __restrict__ kidx, float* __restrict__ kw) {
    __shared__ float sx[SPT], sy[SPT], sz[SPT];
    const int b = blockIdx.y;
    const int tid = threadIdx.x;
    const float* x2 = xyz2 + (size_t)b * SPT * 3;
    for (int s = tid; s < SPT; s += 256) {
        sx[s] = x2[s * 3 + 0];
        sy[s] = x2[s * 3 + 1];
        sz[s] = x2[s * 3 + 2];
    }
    __syncthreads();
    const int n = blockIdx.x * 256 + tid;
    const float* p = xyz1 + ((size_t)b * NPT + n) * 3;
    const float px = p[0], py = p[1], pz = p[2];
    float d0 = FLT_MAX, d1 = FLT_MAX, d2 = FLT_MAX;
    int   i0 = 0,       i1 = 0,       i2 = 0;
    for (int s = 0; s < SPT; ++s) {
        float dx = px - sx[s], dy = py - sy[s], dz = pz - sz[s];
        float d = dx * dx + dy * dy + dz * dz;
        if (d < d2) {
            if (d < d1) {
                d2 = d1; i2 = i1;
                if (d < d0) { d1 = d0; i1 = i0; d0 = d; i0 = s; }
                else        { d1 = d;  i1 = s; }
            } else { d2 = d; i2 = s; }
        }
    }
    float r0 = 1.0f / (d0 + 1e-8f);
    float r1 = 1.0f / (d1 + 1e-8f);
    float r2 = 1.0f / (d2 + 1e-8f);
    float inv = 1.0f / (r0 + r1 + r2);
    size_t base = ((size_t)b * NPT + n) * 3;
    kidx[base + 0] = i0; kidx[base + 1] = i1; kidx[base + 2] = i2;
    kw[base + 0] = r0 * inv; kw[base + 1] = r1 * inv; kw[base + 2] = r2 * inv;
}

// ---------------------------------------------------------------------------
// 2) Build X = concat(points1, interp(points2)) in (B, N, 384) f16 layout
//    (per-point contiguous channels == WMMA B-operand friendly).
//    Block = 32 channel-lanes x 8 points.
// ---------------------------------------------------------------------------
__global__ void buildx_kernel(const float* __restrict__ points1,
                              const float* __restrict__ points2,
                              const int* __restrict__ kidx,
                              const float* __restrict__ kw,
                              _Float16* __restrict__ Xh) {
    const int b  = blockIdx.y;
    const int ty = threadIdx.x >> 5;
    const int tx = threadIdx.x & 31;
    const int n  = blockIdx.x * 8 + ty;
    const size_t kb = ((size_t)b * NPT + n) * 3;
    const int i0 = kidx[kb], i1 = kidx[kb + 1], i2 = kidx[kb + 2];
    const float w0 = kw[kb], w1 = kw[kb + 1], w2 = kw[kb + 2];
    const float* p1 = points1 + (size_t)b * C1_ * NPT;
    const float* p2 = points2 + (size_t)b * C2_ * SPT;
    _Float16* xr = Xh + ((size_t)b * NPT + n) * CIN;
    #pragma unroll
    for (int cc = 0; cc < CIN / 32; ++cc) {
        int c = cc * 32 + tx;
        float v;
        if (c < C1_) {
            v = p1[(size_t)c * NPT + n];
        } else {
            const float* row = p2 + (size_t)(c - C1_) * SPT;
            v = w0 * row[i0] + w1 * row[i1] + w2 * row[i2];
        }
        xr[c] = (_Float16)v;
    }
}

// ---------------------------------------------------------------------------
// 3) Time conditioning: te1 = GELU(t) @ w_t1^T + b_t1  (B x 384)
//                       te2 = GELU(t) @ w_t2^T + b_t2  (B x 256)
// ---------------------------------------------------------------------------
__global__ void te_kernel(const float* __restrict__ t_embed,
                          const float* __restrict__ w_t1, const float* __restrict__ b_t1,
                          const float* __restrict__ w_t2, const float* __restrict__ b_t2,
                          float* __restrict__ te1, float* __restrict__ te2) {
    __shared__ float g[T_];
    const int b = blockIdx.x, j = threadIdx.x;  // blockDim = 384
    if (j < T_) g[j] = gelu_exact(t_embed[b * T_ + j]);
    __syncthreads();
    float s = b_t1[j];
    for (int t = 0; t < T_; ++t) s += w_t1[j * T_ + t] * g[t];
    te1[b * CIN + j] = s;
    if (j < O_) {
        float s2 = b_t2[j];
        for (int t = 0; t < T_; ++t) s2 += w_t2[j * T_ + t] * g[t];
        te2[b * O_ + j] = s2;
    }
}

// 4) Fold te into GEMM bias: bias[b,o] = b_c[o] + sum_c w_c[o,c]*te[b,c]
__global__ void bias_kernel(const float* __restrict__ w_c1, const float* __restrict__ b_c1,
                            const float* __restrict__ te1,
                            const float* __restrict__ w_c2, const float* __restrict__ b_c2,
                            const float* __restrict__ te2,
                            float* __restrict__ bias1, float* __restrict__ bias2) {
    const int b = blockIdx.x, o = threadIdx.x;  // blockDim = 256
    float s1 = b_c1[o];
    for (int c = 0; c < CIN; ++c) s1 += w_c1[o * CIN + c] * te1[b * CIN + c];
    bias1[b * O_ + o] = s1;
    float s2 = b_c2[o];
    for (int c = 0; c < O_; ++c) s2 += w_c2[o * O_ + c] * te2[b * O_ + c];
    bias2[b * O_ + o] = s2;
}

// 5) Convert conv weights to f16 (row-major (O, K) already WMMA-A friendly)
__global__ void cvtw_kernel(const float* __restrict__ w_c1, const float* __restrict__ w_c2,
                            _Float16* __restrict__ Wh1, _Float16* __restrict__ Wh2) {
    int idx = blockIdx.x * 256 + threadIdx.x;
    if (idx < O_ * CIN) Wh1[idx] = (_Float16)w_c1[idx];
    int j = idx - O_ * CIN;
    if (j >= 0 && j < O_ * O_) Wh2[j] = (_Float16)w_c2[j];
}

// ---------------------------------------------------------------------------
// 6) WMMA GEMM: Y[b,n,o] = sum_c Wh[o,c]*Xh[b,n,c] + bias[b,o]
//    Block = 256 threads = 8 waves; wave -> one 16(o) x 16(n) tile.
//    16-bit operand layout: lane<16 holds K+0..7 / K+16..23, lane>=16 holds
//    K+8..15 / K+24..31 -> two 16-byte loads per 32-K step per operand.
// ---------------------------------------------------------------------------
template <int K>
__global__ void gemm_kernel(const _Float16* __restrict__ Xh,
                            const _Float16* __restrict__ Wh,
                            const float* __restrict__ bias,
                            float* __restrict__ Y) {
    const int lane = threadIdx.x & 31;
    const int wave = threadIdx.x >> 5;
    const int half = lane >> 4;
    const int lr   = lane & 15;
    const int b      = blockIdx.z;
    const int o_base = blockIdx.y * 128 + wave * 16;
    const int n_base = blockIdx.x * 16;

    const _Float16* arow = Wh + (size_t)(o_base + lr) * K + half * 8;
    const _Float16* brow = Xh + ((size_t)b * NPT + n_base + lr) * K + half * 8;

    v8f acc = {};
    #pragma unroll
    for (int k = 0; k < K; k += 32) {
        v8h alo = *(const v8h*)(arow + k);
        v8h ahi = *(const v8h*)(arow + k + 16);
        v8h blo = *(const v8h*)(brow + k);
        v8h bhi = *(const v8h*)(brow + k + 16);
        v16h a  = __builtin_shufflevector(alo, ahi, 0,1,2,3,4,5,6,7,8,9,10,11,12,13,14,15);
        v16h bm = __builtin_shufflevector(blo, bhi, 0,1,2,3,4,5,6,7,8,9,10,11,12,13,14,15);
        acc = __builtin_amdgcn_wmma_f32_16x16x32_f16(
            /*neg_a=*/false, a, /*neg_b=*/false, bm,
            /*c_mod=*/(short)0, acc, /*reuse_a=*/false, /*reuse_b=*/false);
    }

    // f32 C/D layout: VGPR r -> o = o_base + half*8 + r, n = n_base + lr
    const int o0 = o_base + half * 8;
    const int n  = n_base + lr;
    float* yp = Y + ((size_t)b * NPT + n) * O_ + o0;
    const float* bp = bias + b * O_ + o0;
    #pragma unroll
    for (int r = 0; r < 8; ++r) yp[r] = acc[r] + bp[r];
}

// ---------------------------------------------------------------------------
// 7) BN stats: deterministic two-stage reduction over 65536 rows x 256 chans.
// ---------------------------------------------------------------------------
__global__ void stats_kernel(const float* __restrict__ Y,
                             float* __restrict__ psum, float* __restrict__ psq) {
    const int t = threadIdx.x;            // channel
    const int blk = blockIdx.x;           // 256 blocks x 256 rows each
    float s = 0.f, sq = 0.f;
    size_t base = (size_t)blk * 256 * O_;
    for (int r = 0; r < 256; ++r) {
        float v = Y[base + (size_t)r * O_ + t];
        s += v; sq += v * v;
    }
    psum[blk * O_ + t] = s;
    psq [blk * O_ + t] = sq;
}

__global__ void finalize_kernel(const float* __restrict__ psum, const float* __restrict__ psq,
                                const float* __restrict__ gamma, const float* __restrict__ beta,
                                float* __restrict__ scale, float* __restrict__ shift) {
    const int o = threadIdx.x;            // 1 block, 256 threads
    float s = 0.f, sq = 0.f;
    for (int p = 0; p < 256; ++p) { s += psum[p * O_ + o]; sq += psq[p * O_ + o]; }
    const float invM = 1.0f / (float)ROWS;
    float mean = s * invM;
    float var  = sq * invM - mean * mean;
    float sc = gamma[o] * rsqrtf(var + 1e-5f);
    scale[o] = sc;
    shift[o] = beta[o] - mean * sc;
}

// 8) BN + GELU, re-quantize to f16 for GEMM-2 input (same (B,N,O) layout)
__global__ void bngelu_h_kernel(const float* __restrict__ Y,
                                const float* __restrict__ scale,
                                const float* __restrict__ shift,
                                _Float16* __restrict__ Xo) {
    size_t idx = (size_t)blockIdx.x * 256 + threadIdx.x;
    int o = (int)(idx & (O_ - 1));
    float v = Y[idx];
    Xo[idx] = (_Float16)gelu_exact(scale[o] * v + shift[o]);
}

// 9) Final BN + GELU with 16x16 LDS transpose to o-major output (B, 256, N)
__global__ void bngelu_out_kernel(const float* __restrict__ Y,
                                  const float* __restrict__ scale,
                                  const float* __restrict__ shift,
                                  float* __restrict__ out) {
    __shared__ float tile[16][17];
    const int b = blockIdx.z;
    const int n0 = blockIdx.x * 16, o0 = blockIdx.y * 16;
    const int tx = threadIdx.x, ty = threadIdx.y;
    tile[ty][tx] = Y[((size_t)b * NPT + n0 + ty) * O_ + o0 + tx];
    __syncthreads();
    const int o = o0 + ty;
    float v = scale[o] * tile[tx][ty] + shift[o];
    out[((size_t)b * O_ + o) * NPT + n0 + tx] = gelu_exact(v);
}

// ---------------------------------------------------------------------------
extern "C" void kernel_launch(void* const* d_in, const int* in_sizes, int n_in,
                              void* d_out, int out_size, void* d_ws, size_t ws_size,
                              hipStream_t stream) {
    (void)in_sizes; (void)n_in; (void)out_size; (void)ws_size;
    const float* xyz1    = (const float*)d_in[0];
    const float* xyz2    = (const float*)d_in[1];
    const float* points1 = (const float*)d_in[2];
    const float* points2 = (const float*)d_in[3];
    const float* t_embed = (const float*)d_in[4];
    const float* w_t1 = (const float*)d_in[5];
    const float* b_t1 = (const float*)d_in[6];
    const float* w_c1 = (const float*)d_in[7];
    const float* b_c1 = (const float*)d_in[8];
    const float* g1   = (const float*)d_in[9];
    const float* be1  = (const float*)d_in[10];
    const float* w_t2 = (const float*)d_in[11];
    const float* b_t2 = (const float*)d_in[12];
    const float* w_c2 = (const float*)d_in[13];
    const float* b_c2 = (const float*)d_in[14];
    const float* g2   = (const float*)d_in[15];
    const float* be2  = (const float*)d_in[16];
    float* out = (float*)d_out;

    char* ws = (char*)d_ws;
    size_t off = 0;
    auto alloc = [&](size_t bytes) -> void* {
        void* p = ws + off;
        off = (off + bytes + 255) & ~(size_t)255;
        return p;
    };
    _Float16* Xh  = (_Float16*)alloc((size_t)ROWS * CIN * sizeof(_Float16)); // 48MB
    float*    Y   = (float*)   alloc((size_t)ROWS * O_ * sizeof(float));     // 64MB
    int*      kidx= (int*)     alloc((size_t)ROWS * 3 * sizeof(int));
    float*    kw  = (float*)   alloc((size_t)ROWS * 3 * sizeof(float));
    _Float16* Wh1 = (_Float16*)alloc((size_t)O_ * CIN * sizeof(_Float16));
    _Float16* Wh2 = (_Float16*)alloc((size_t)O_ * O_ * sizeof(_Float16));
    float*    te1 = (float*)   alloc((size_t)B_ * CIN * sizeof(float));
    float*    te2 = (float*)   alloc((size_t)B_ * O_ * sizeof(float));
    float*    bias1=(float*)   alloc((size_t)B_ * O_ * sizeof(float));
    float*    bias2=(float*)   alloc((size_t)B_ * O_ * sizeof(float));
    float*    psum= (float*)   alloc((size_t)256 * O_ * sizeof(float));
    float*    psq = (float*)   alloc((size_t)256 * O_ * sizeof(float));
    float*    sc1 = (float*)   alloc(O_ * sizeof(float));
    float*    sh1 = (float*)   alloc(O_ * sizeof(float));
    float*    sc2 = (float*)   alloc(O_ * sizeof(float));
    float*    sh2 = (float*)   alloc(O_ * sizeof(float));
    _Float16* X2h = Xh;  // reuse: 32MB <= 48MB, Y1 fully consumed before GEMM-2

    cvtw_kernel<<<(O_ * CIN + O_ * O_ + 255) / 256, 256, 0, stream>>>(w_c1, w_c2, Wh1, Wh2);
    te_kernel<<<B_, 384, 0, stream>>>(t_embed, w_t1, b_t1, w_t2, b_t2, te1, te2);
    bias_kernel<<<B_, 256, 0, stream>>>(w_c1, b_c1, te1, w_c2, b_c2, te2, bias1, bias2);
    knn_kernel<<<dim3(NPT / 256, B_), 256, 0, stream>>>(xyz1, xyz2, kidx, kw);
    buildx_kernel<<<dim3(NPT / 8, B_), 256, 0, stream>>>(points1, points2, kidx, kw, Xh);

    // Block 1: 384 -> 256
    gemm_kernel<CIN><<<dim3(NPT / 16, 2, B_), 256, 0, stream>>>(Xh, Wh1, bias1, Y);
    stats_kernel<<<256, 256, 0, stream>>>(Y, psum, psq);
    finalize_kernel<<<1, 256, 0, stream>>>(psum, psq, g1, be1, sc1, sh1);
    bngelu_h_kernel<<<(int)((size_t)ROWS * O_ / 256), 256, 0, stream>>>(Y, sc1, sh1, X2h);

    // Block 2: 256 -> 256
    gemm_kernel<O_><<<dim3(NPT / 16, 2, B_), 256, 0, stream>>>(X2h, Wh2, bias2, Y);
    stats_kernel<<<256, 256, 0, stream>>>(Y, psum, psq);
    finalize_kernel<<<1, 256, 0, stream>>>(psum, psq, g2, be2, sc2, sh2);
    bngelu_out_kernel<<<dim3(NPT / 16, O_ / 16, B_), dim3(16, 16), 0, stream>>>(Y, sc2, sh2, out);
}
